// Qwen3MoeAttention_34720515620974
// MI455X (gfx1250) — matrix operand order repaired and checked by reading, not compiled, not run on hardware
//
#include <hip/hip_runtime.h>
#include <hip/hip_bf16.h>

// ---------------------------------------------------------------------------
// Qwen3-style attention layer for MI455X (gfx1250), bf16 WMMA everywhere.
// B=2 S=2048 HID=2048 H=32 KVH=4 D=128.  Compute-bound (~290 GFLOP vs ~110MB).
// GEMMs: 128x128 block tiles double-buffered in LDS via gfx1250 async
// global->LDS copies (ASYNCcnt), 8 waves x (32x64) WMMA register tiles.
// ---------------------------------------------------------------------------

typedef __attribute__((ext_vector_type(16))) __bf16 v16bf;
typedef __attribute__((ext_vector_type(8)))  float  v8f;

constexpr int B_ = 2, S_ = 2048, HID_ = 2048, H_ = 32, KVH_ = 4, D_ = 128;
constexpr float EPS_   = 1e-6f;
constexpr float SCALE_ = 0.08838834764831845f;   // 1/sqrt(128)
constexpr float NEG_   = -3.0e38f;

__device__ __forceinline__ __bf16 bf16_rn(float f) {
  union { float f; unsigned u; } a; a.f = f;
  unsigned u = a.u + 0x7FFFu + ((a.u >> 16) & 1u);   // round-to-nearest-even
  union { unsigned short s; __bf16 b; } r; r.s = (unsigned short)(u >> 16);
  return r.b;
}

__device__ __forceinline__ v8f zero8() {
  v8f z = {0.f, 0.f, 0.f, 0.f, 0.f, 0.f, 0.f, 0.f};
  return z;
}

// -------------------- CDNA5 async global->LDS helpers ----------------------
// LDS offset of a __shared__ object = low 32 bits of its generic address
// (LDS aperture: addr[31:0] is the wave-relative LDS byte address).
__device__ __forceinline__ unsigned lds_off(const void* p) {
  return (unsigned)(size_t)p;
}

// GLOBAL_LOAD_ASYNC_TO_LDS_B128: LDS[vdst] = MEM[vaddr], tracked by ASYNCcnt.
__device__ __forceinline__ void async_copy_b128(unsigned dst_lds, const void* src) {
  asm volatile("global_load_async_to_lds_b128 %0, %1, off"
               :: "v"(dst_lds), "v"(src)
               : "memory");
}

template <int N>
__device__ __forceinline__ void wait_async() {
#if __has_builtin(__builtin_amdgcn_s_wait_asynccnt)
  __builtin_amdgcn_s_wait_asynccnt(N);
#else
  asm volatile("s_wait_asynccnt %0" :: "i"(N) : "memory");
#endif
}

// -------------------- WMMA fragment loaders (ISA 7.12.2) -------------------
// A-matrix fragment (16x32, 16-bit): lane r=lane%16 selects row;
//   VGPR v<4 : K = hi*8 + 2v,2v+1   VGPR v>=4 : K = 16 + hi*8 + 2(v-4),+1
__device__ __forceinline__ v16bf load_fragA(const __bf16* rowptr, int kbase, int hi) {
  union { v16bf v; unsigned u[8]; } r;
  const __bf16* p = rowptr + kbase + hi * 8;
  r.u[0] = *reinterpret_cast<const unsigned*>(p + 0);
  r.u[1] = *reinterpret_cast<const unsigned*>(p + 2);
  r.u[2] = *reinterpret_cast<const unsigned*>(p + 4);
  r.u[3] = *reinterpret_cast<const unsigned*>(p + 6);
  r.u[4] = *reinterpret_cast<const unsigned*>(p + 16);
  r.u[5] = *reinterpret_cast<const unsigned*>(p + 18);
  r.u[6] = *reinterpret_cast<const unsigned*>(p + 20);
  r.u[7] = *reinterpret_cast<const unsigned*>(p + 22);
  return r.v;
}

// B-matrix fragment (32x16, 16-bit): lane r=lane%16 selects column n;
//   VGPR v<4 : K = hi*8 + 2v,2v+1   VGPR v>=4 : K = 16 + hi*8 + 2(v-4),+1
// rowptr points at B^T row n (K-contiguous).
__device__ __forceinline__ v16bf load_fragB(const __bf16* rowptr, int kbase, int hi) {
  union { v16bf v; unsigned u[8]; } r;
  const __bf16* p = rowptr + kbase + hi * 8;
  r.u[0] = *reinterpret_cast<const unsigned*>(p + 0);
  r.u[1] = *reinterpret_cast<const unsigned*>(p + 2);
  r.u[2] = *reinterpret_cast<const unsigned*>(p + 4);
  r.u[3] = *reinterpret_cast<const unsigned*>(p + 6);
  r.u[4] = *reinterpret_cast<const unsigned*>(p + 16);
  r.u[5] = *reinterpret_cast<const unsigned*>(p + 18);
  r.u[6] = *reinterpret_cast<const unsigned*>(p + 20);
  r.u[7] = *reinterpret_cast<const unsigned*>(p + 22);
  return r.v;
}

__device__ __forceinline__ v8f wmma_bf16(v16bf a, v16bf b, v8f c) {
  return __builtin_amdgcn_wmma_f32_16x16x32_bf16(false, a, false, b, (short)0, c,
                                                 false, false);
}

// ---------------------------------------------------------------------------
// elementwise fp32 -> bf16
__global__ void f32_to_bf16_kern(const float* __restrict__ in,
                                 __bf16* __restrict__ out, long n) {
  long i = (long)blockIdx.x * blockDim.x + threadIdx.x;
  long stride = (long)gridDim.x * blockDim.x;
  for (; i < n; i += stride) out[i] = bf16_rn(in[i]);
}

// fp32 (K x N) -> bf16 (N x K)  (weight transpose so GEMM is TN / K-contiguous)
__global__ void transpose_convert_kern(const float* __restrict__ in,
                                       __bf16* __restrict__ out, int K, int N) {
  __shared__ float tile[32][33];
  const int k0 = blockIdx.x * 32, n0 = blockIdx.y * 32;
  const int tx = threadIdx.x, ty = threadIdx.y;   // 32 x 8
  #pragma unroll
  for (int i = 0; i < 32; i += 8)
    tile[ty + i][tx] = in[(size_t)(k0 + ty + i) * N + n0 + tx];
  __syncthreads();
  #pragma unroll
  for (int i = 0; i < 32; i += 8)
    out[(size_t)(n0 + ty + i) * K + k0 + tx] = bf16_rn(tile[tx][ty + i]);
}

// ---------------------------------------------------------------------------
// C(MxN, f32) = A(MxK, bf16 row-major) @ Bt(NxK, bf16 row-major)^T
// Block: 256 threads / 8 waves -> 128x128 tile.  Per 32-deep K-step the block
// async-copies a 128x32 A slice and a 128x32 Bt slice into LDS (double
// buffered, ASYNCcnt-tracked) while WMMAs consume the previous slice.
// Each wave: 32x64 register tile = 8 WMMAs per K-step from LDS.
__global__ void __launch_bounds__(256) gemm_bf16_tn_lds(
    float* __restrict__ C, const __bf16* __restrict__ A,
    const __bf16* __restrict__ Bt, int M, int N, int K) {
  __shared__ __align__(16) __bf16 As[2][128 * 32];
  __shared__ __align__(16) __bf16 Bs[2][128 * 32];

  const int tid  = threadIdx.x;
  const int lane = tid & 31, wave = tid >> 5;
  const int r = lane & 15, hi = lane >> 4;
  const int mi = wave & 3;        // 4 waves stacked in M (32 rows each)
  const int ni = wave >> 2;       // 2 waves stacked in N (64 cols each)
  const long bm = (long)blockIdx.x * 128;
  const long bn = (long)blockIdx.y * 128;

  // staging assignment: 512 16-byte chunks per 128x32 tile; thread copies
  // chunks tid and tid+256 of A and of B.
  const int c0 = tid, c1 = tid + 256;
  const int ar0 = c0 >> 2, ao0 = c0 & 3;
  const int ar1 = c1 >> 2, ao1 = c1 & 3;
  const __bf16* Ag = A + bm * K;
  const __bf16* Bg = Bt + bn * K;
  const unsigned aB[2] = { lds_off(&As[0][0]), lds_off(&As[1][0]) };
  const unsigned bB[2] = { lds_off(&Bs[0][0]), lds_off(&Bs[1][0]) };

  auto stage = [&](int kk, int buf) {
    async_copy_b128(aB[buf] + ar0 * 64 + ao0 * 16, Ag + (long)ar0 * K + kk + ao0 * 8);
    async_copy_b128(aB[buf] + ar1 * 64 + ao1 * 16, Ag + (long)ar1 * K + kk + ao1 * 8);
    async_copy_b128(bB[buf] + ar0 * 64 + ao0 * 16, Bg + (long)ar0 * K + kk + ao0 * 8);
    async_copy_b128(bB[buf] + ar1 * 64 + ao1 * 16, Bg + (long)ar1 * K + kk + ao1 * 8);
  };

  v8f acc[8];
  #pragma unroll
  for (int i = 0; i < 8; ++i) acc[i] = zero8();

  auto compute = [&](int buf) {
    const __bf16* as = &As[buf][0];
    const __bf16* bs = &Bs[buf][0];
    v16bf a0 = load_fragA(as + (mi * 32 + r) * 32, 0, hi);
    v16bf a1 = load_fragA(as + (mi * 32 + 16 + r) * 32, 0, hi);
    #pragma unroll
    for (int j = 0; j < 4; ++j) {
      v16bf bf = load_fragB(bs + (ni * 64 + j * 16 + r) * 32, 0, hi);
      acc[j]     = wmma_bf16(a0, bf, acc[j]);
      acc[4 + j] = wmma_bf16(a1, bf, acc[4 + j]);
    }
  };

  stage(0, 0);
  int buf = 0;
  int k = 0;
  for (; k < K - 32; k += 32) {
    stage(k + 32, buf ^ 1);   // newest 4 asyncs stay in flight
    wait_async<4>();          // oldest 4 (current buffer) retired
    __syncthreads();
    compute(buf);
    __syncthreads();
    buf ^= 1;
  }
  wait_async<0>();
  __syncthreads();
  compute(buf);

  #pragma unroll
  for (int p = 0; p < 2; ++p) {
    float* crow = C + (bm + mi * 32 + p * 16 + hi * 8) * N + bn + ni * 64 + r;
    #pragma unroll
    for (int v = 0; v < 8; ++v) {
      crow[(long)v * N +  0] = acc[p * 4 + 0][v];
      crow[(long)v * N + 16] = acc[p * 4 + 1][v];
      crow[(long)v * N + 32] = acc[p * 4 + 2][v];
      crow[(long)v * N + 48] = acc[p * 4 + 3][v];
    }
  }
}

// ---------------------------------------------------------------------------
// Fused RMSNorm + RoPE per head-token; one wave handles one 128-wide vector.
// X:(B*S, Hn*D) f32 -> out:(B,Hn,S,D) bf16.  rotate-half partner = shfl_xor 16.
__global__ void __launch_bounds__(256) qk_norm_rope_kern(
    const float* __restrict__ X, const float* __restrict__ cs,
    const float* __restrict__ sn, const float* __restrict__ w,
    __bf16* __restrict__ out, int Hn, long ntask) {
  const int lane = threadIdx.x & 31;
  long wid = (((long)blockIdx.x * blockDim.x) + threadIdx.x) >> 5;
  if (wid >= ntask) return;
  const long bs = wid / Hn;
  const int hh = (int)(wid % Hn);
  const float* xp = X + (bs * Hn + hh) * (long)D_ + lane * 4;
  float x0 = xp[0], x1 = xp[1], x2 = xp[2], x3 = xp[3];
  float ss = x0 * x0 + x1 * x1 + x2 * x2 + x3 * x3;
  ss += __shfl_xor(ss, 1);  ss += __shfl_xor(ss, 2);
  ss += __shfl_xor(ss, 4);  ss += __shfl_xor(ss, 8);
  ss += __shfl_xor(ss, 16);
  const float inv = rsqrtf(ss * (1.0f / (float)D_) + EPS_);
  const float* wp = w + lane * 4;
  float n0 = x0 * inv * wp[0], n1 = x1 * inv * wp[1];
  float n2 = x2 * inv * wp[2], n3 = x3 * inv * wp[3];
  float o0 = __shfl_xor(n0, 16), o1 = __shfl_xor(n1, 16);
  float o2 = __shfl_xor(n2, 16), o3 = __shfl_xor(n3, 16);
  const float* cp = cs + bs * (long)D_ + lane * 4;
  const float* sp = sn + bs * (long)D_ + lane * 4;
  const float sg = (lane < 16) ? -1.0f : 1.0f;
  float y0 = n0 * cp[0] + sg * o0 * sp[0];
  float y1 = n1 * cp[1] + sg * o1 * sp[1];
  float y2 = n2 * cp[2] + sg * o2 * sp[2];
  float y3 = n3 * cp[3] + sg * o3 * sp[3];
  const long b = bs / S_, si = bs % S_;
  __bf16* op = out + (((b * Hn + hh) * (long)S_) + si) * D_ + lane * 4;
  op[0] = bf16_rn(y0); op[1] = bf16_rn(y1);
  op[2] = bf16_rn(y2); op[3] = bf16_rn(y3);
}

// V (B,S,KVH*D) f32 -> Vt (B,KVH,D,S) bf16  (so PV B-frags are S-contiguous)
__global__ void v_transpose_kern(const float* __restrict__ Vf,
                                 __bf16* __restrict__ Vt) {
  const long total = (long)B_ * KVH_ * D_ * S_;
  long i = (long)blockIdx.x * blockDim.x + threadIdx.x;
  long stride = (long)gridDim.x * blockDim.x;
  for (; i < total; i += stride) {
    const int  s   = (int)(i & (S_ - 1));
    long t = i >> 11;                 // /S_
    const int  d   = (int)(t & (D_ - 1));
    t >>= 7;                          // /D_
    const int  kvh = (int)(t & (KVH_ - 1));
    const long b   = t >> 2;
    Vt[i] = bf16_rn(Vf[((b * S_ + s) * KVH_ + kvh) * (long)D_ + d]);
  }
}

// ---------------------------------------------------------------------------
// Flash attention: one wave per 16 query rows of one (b,h).  32-key blocks:
// 8 WMMAs for QK^T, 8 for PV, online softmax in fp32, P staged via LDS to
// convert C-layout -> A-layout.  Causal: only key blocks up to the diagonal.
__global__ void __launch_bounds__(128) flash_attn_kern(
    const __bf16* __restrict__ Qb,   // (B,H,S,D)
    const __bf16* __restrict__ Kb,   // (B,KVH,S,D)
    const __bf16* __restrict__ Vt,   // (B,KVH,D,S)
    __bf16* __restrict__ Ob) {       // (B,S,H,D)
  __shared__ __bf16 Pl[4][16 * 32];
  const int lane = threadIdx.x & 31;
  const int wave = threadIdx.x >> 5;
  const int r = lane & 15, hi = lane >> 4;
  const int QT = S_ / 16;
  long task = (long)blockIdx.x * 4 + wave;
  if (task >= (long)B_ * H_ * QT) return;
  const int qt = (int)(task % QT);
  const int bh = (int)(task / QT);
  const int h = bh % H_, b = bh / H_;
  const int kvh = h / (H_ / KVH_);
  const int q0 = qt * 16;

  const __bf16* qrow = Qb + (((long)(b * H_ + h) * S_) + q0 + r) * D_;
  v16bf qf0 = load_fragA(qrow,  0, hi);
  v16bf qf1 = load_fragA(qrow, 32, hi);
  v16bf qf2 = load_fragA(qrow, 64, hi);
  v16bf qf3 = load_fragA(qrow, 96, hi);

  const __bf16* kb = Kb + ((long)(b * KVH_ + kvh) * S_) * D_;
  const __bf16* vb = Vt + ((long)(b * KVH_ + kvh) * D_) * S_;

  v8f acc[8];
  float mi[8], li[8];
  #pragma unroll
  for (int v = 0; v < 8; ++v) { acc[v] = zero8(); mi[v] = NEG_; li[v] = 0.0f; }

  const int kend = q0 + 16;                 // last needed key is q0+15
  for (int k0 = 0; k0 < kend; k0 += 32) {
    const bool have1 = (k0 + 16) < kend;
    v8f s0 = zero8(), s1 = zero8();
    const __bf16* k0row = kb + (long)(k0 + r) * D_;
    s0 = wmma_bf16(qf0, load_fragB(k0row,  0, hi), s0);
    s0 = wmma_bf16(qf1, load_fragB(k0row, 32, hi), s0);
    s0 = wmma_bf16(qf2, load_fragB(k0row, 64, hi), s0);
    s0 = wmma_bf16(qf3, load_fragB(k0row, 96, hi), s0);
    if (have1) {
      const __bf16* k1row = k0row + 16L * D_;
      s1 = wmma_bf16(qf0, load_fragB(k1row,  0, hi), s1);
      s1 = wmma_bf16(qf1, load_fragB(k1row, 32, hi), s1);
      s1 = wmma_bf16(qf2, load_fragB(k1row, 64, hi), s1);
      s1 = wmma_bf16(qf3, load_fragB(k1row, 96, hi), s1);
    }

    float mnew[8];
    #pragma unroll
    for (int v = 0; v < 8; ++v) {
      const int q = q0 + v + hi * 8;
      float a = s0[v] * SCALE_;
      if (k0 + r > q) a = NEG_;
      float c = have1 ? s1[v] * SCALE_ : NEG_;
      if (k0 + 16 + r > q) c = NEG_;
      s0[v] = a; s1[v] = c;
      float t = fmaxf(a, c);
      t = fmaxf(t, __shfl_xor(t, 1));
      t = fmaxf(t, __shfl_xor(t, 2));
      t = fmaxf(t, __shfl_xor(t, 4));
      t = fmaxf(t, __shfl_xor(t, 8));
      mnew[v] = fmaxf(mi[v], t);
    }

    #pragma unroll
    for (int v = 0; v < 8; ++v) {
      const float p0 = __expf(s0[v] - mnew[v]);
      const float p1 = __expf(s1[v] - mnew[v]);
      const float rs = __expf(mi[v] - mnew[v]);
      float ps = p0 + p1;
      ps += __shfl_xor(ps, 1);
      ps += __shfl_xor(ps, 2);
      ps += __shfl_xor(ps, 4);
      ps += __shfl_xor(ps, 8);
      li[v] = li[v] * rs + ps;
      mi[v] = mnew[v];
      #pragma unroll
      for (int dc = 0; dc < 8; ++dc) acc[dc][v] *= rs;
      Pl[wave][(v + hi * 8) * 32 + r]      = bf16_rn(p0);
      Pl[wave][(v + hi * 8) * 32 + 16 + r] = bf16_rn(p1);
    }

    v16bf pf = load_fragA(&Pl[wave][r * 32], 0, hi);
    #pragma unroll
    for (int dc = 0; dc < 8; ++dc) {
      const __bf16* vrow = vb + (long)(dc * 16 + r) * S_;
      acc[dc] = wmma_bf16(pf, load_fragB(vrow, k0, hi), acc[dc]);
    }
  }

  #pragma unroll
  for (int v = 0; v < 8; ++v) {
    const float inv = 1.0f / li[v];
    const long orow = ((long)(b * S_ + q0 + v + hi * 8) * H_ + h) * D_;
    #pragma unroll
    for (int dc = 0; dc < 8; ++dc)
      Ob[orow + dc * 16 + r] = bf16_rn(acc[dc][v] * inv);
  }
}

// ---------------------------------------------------------------------------
extern "C" void kernel_launch(void* const* d_in, const int* in_sizes, int n_in,
                              void* d_out, int out_size, void* d_ws, size_t ws_size,
                              hipStream_t stream) {
  const float* hidden = (const float*)d_in[0];
  const float* cosb   = (const float*)d_in[1];
  const float* sinb   = (const float*)d_in[2];
  const float* q_w    = (const float*)d_in[3];
  const float* k_w    = (const float*)d_in[4];
  const float* v_w    = (const float*)d_in[5];
  const float* o_w    = (const float*)d_in[6];
  const float* q_nw   = (const float*)d_in[7];
  const float* k_nw   = (const float*)d_in[8];
  // d_in[9] = causal mask: structurally known, handled in-kernel.
  float* out = (float*)d_out;

  char* ws = (char*)d_ws;
  size_t off = 0;
  auto take = [&](size_t bytes) -> char* {
    char* p = ws + off;
    off += (bytes + 255) & ~(size_t)255;
    return p;
  };

  const long MT = (long)B_ * S_;   // 4096 token rows
  __bf16* Xb  = (__bf16*)take((size_t)MT * HID_ * 2);
  __bf16* Wqt = (__bf16*)take((size_t)(H_ * D_) * HID_ * 2);
  __bf16* Wkt = (__bf16*)take((size_t)(KVH_ * D_) * HID_ * 2);
  __bf16* Wvt = (__bf16*)take((size_t)(KVH_ * D_) * HID_ * 2);
  __bf16* Wot = (__bf16*)take((size_t)HID_ * (H_ * D_) * 2);
  float*  Qf  = (float*)take((size_t)MT * (H_ * D_) * 4);
  float*  Kf  = (float*)take((size_t)MT * (KVH_ * D_) * 4);
  float*  Vf  = (float*)take((size_t)MT * (KVH_ * D_) * 4);
  __bf16* Qbf = (__bf16*)take((size_t)B_ * H_ * S_ * D_ * 2);
  __bf16* Kbf = (__bf16*)take((size_t)B_ * KVH_ * S_ * D_ * 2);
  __bf16* Vtb = (__bf16*)take((size_t)B_ * KVH_ * D_ * S_ * 2);
  __bf16* Ab  = (__bf16*)Qf;   // attention output reuses Qf region (consumed)

  // 1) hidden -> bf16
  f32_to_bf16_kern<<<4096, 256, 0, stream>>>(hidden, Xb, MT * HID_);
  // 2) weights -> bf16, transposed to N x K
  transpose_convert_kern<<<dim3(HID_ / 32, (H_ * D_) / 32), dim3(32, 8), 0, stream>>>(
      q_w, Wqt, HID_, H_ * D_);
  transpose_convert_kern<<<dim3(HID_ / 32, (KVH_ * D_) / 32), dim3(32, 8), 0, stream>>>(
      k_w, Wkt, HID_, KVH_ * D_);
  transpose_convert_kern<<<dim3(HID_ / 32, (KVH_ * D_) / 32), dim3(32, 8), 0, stream>>>(
      v_w, Wvt, HID_, KVH_ * D_);
  transpose_convert_kern<<<dim3((H_ * D_) / 32, HID_ / 32), dim3(32, 8), 0, stream>>>(
      o_w, Wot, H_ * D_, HID_);

  // 3) QKV projections (WMMA + async-LDS double buffering)
  gemm_bf16_tn_lds<<<dim3(MT / 128, (H_ * D_) / 128), 256, 0, stream>>>(
      Qf, Xb, Wqt, (int)MT, H_ * D_, HID_);
  gemm_bf16_tn_lds<<<dim3(MT / 128, (KVH_ * D_) / 128), 256, 0, stream>>>(
      Kf, Xb, Wkt, (int)MT, KVH_ * D_, HID_);
  gemm_bf16_tn_lds<<<dim3(MT / 128, (KVH_ * D_) / 128), 256, 0, stream>>>(
      Vf, Xb, Wvt, (int)MT, KVH_ * D_, HID_);

  // 4) RMSNorm + RoPE -> (B,Hn,S,D) bf16
  const long qtasks = MT * H_;
  qk_norm_rope_kern<<<(int)((qtasks + 7) / 8), 256, 0, stream>>>(
      Qf, cosb, sinb, q_nw, Qbf, H_, qtasks);
  const long ktasks = MT * KVH_;
  qk_norm_rope_kern<<<(int)((ktasks + 7) / 8), 256, 0, stream>>>(
      Kf, cosb, sinb, k_nw, Kbf, KVH_, ktasks);
  // 5) V -> (B,KVH,D,S) bf16
  v_transpose_kern<<<2048, 256, 0, stream>>>(Vf, Vtb);

  // 6) causal flash attention (WMMA) -> (B,S,H,D) bf16
  flash_attn_kern<<<(int)(((long)B_ * H_ * (S_ / 16) + 3) / 4), 128, 0, stream>>>(
      Qbf, Kbf, Vtb, Ab);

  // 7) output projection (WMMA + async-LDS) -> fp32 d_out
  gemm_bf16_tn_lds<<<dim3(MT / 128, HID_ / 128), 256, 0, stream>>>(
      out, Ab, Wot, (int)MT, HID_, H_ * D_);

  (void)in_sizes; (void)n_in; (void)out_size; (void)ws_size;
}
